// MultiHeadFrequencyCrossAttention_84404697301164
// MI455X (gfx1250) — matrix operand order repaired and checked by reading, not compiled, not run on hardware
//
#include <hip/hip_runtime.h>
#include <hip/hip_bf16.h>

typedef __bf16 bf16;
typedef __attribute__((ext_vector_type(16))) __bf16 v16bf;
typedef __attribute__((ext_vector_type(8)))  __bf16 v8bf;
typedef __attribute__((ext_vector_type(4)))  __bf16 v4bf;
typedef __attribute__((ext_vector_type(8)))  float  v8f;
typedef __attribute__((ext_vector_type(4)))  float  v4f;

#define B_  4
#define T_  1024
#define E_  512
#define H_  8
#define DH_ 64
// dk * D = sqrt(64) * 64 = 512 : scale from the collapsed FFT identity
#define SCALE_ 512.0f

// ---- WMMA fragment loaders from LDS (bf16) ------------------------------
// A operand (16x32, MxK), LDS tile stored row-major [m][k], pitch multiple of 8.
__device__ __forceinline__ v16bf frag_A(const bf16* p0, int pitch, int m0, int k0, int lane) {
  const bf16* p = p0 + (m0 + (lane & 15)) * pitch + k0 + ((lane >> 4) << 3);
  v8bf lo = *(const v8bf*)p;
  v8bf hi = *(const v8bf*)(p + 16);
  v16bf a;
#pragma unroll
  for (int i = 0; i < 8; ++i) { a[i] = lo[i]; a[i + 8] = hi[i]; }
  return a;
}

// B operand (32x16, KxN), LDS tile stored TRANSPOSED [n][k], pitch multiple of 8.
__device__ __forceinline__ v16bf frag_B(const bf16* p0, int pitch, int n0, int k0, int lane) {
  const bf16* p = p0 + (n0 + (lane & 15)) * pitch + k0 + ((lane >> 4) << 3);
  v8bf lo = *(const v8bf*)p;
  v8bf hi = *(const v8bf*)(p + 16);
  v16bf b;
#pragma unroll
  for (int i = 0; i < 8; ++i) { b[i] = lo[i]; b[i + 8] = hi[i]; }
  return b;
}

#define WMMA_BF16(A, B, C) \
  __builtin_amdgcn_wmma_f32_16x16x32_bf16(false, (A), false, (B), (short)0, (C), false, false)

// ---- staging helpers (fully unrolled so loads clause) --------------------
__device__ __forceinline__ void stage_xw(const float* __restrict__ X,
                                         const float* __restrict__ W,
                                         bf16 (*Xs)[40], bf16 (*Wt)[40],
                                         int tile_m, int tile_n, int k0, int tid) {
#pragma unroll
  for (int it = 0; it < 4; ++it) {
    int idx = tid + it * 128;
    int r = idx >> 3, c4 = (idx & 7) << 2;
    v4f f = *(const v4f*)&X[(tile_m + r) * E_ + k0 + c4];
    v4bf p; p[0] = (bf16)f[0]; p[1] = (bf16)f[1]; p[2] = (bf16)f[2]; p[3] = (bf16)f[3];
    *(v4bf*)&Xs[r][c4] = p;
  }
#pragma unroll
  for (int it = 0; it < 4; ++it) {
    int idx = tid + it * 128;
    int kk = idx >> 4, n4 = (idx & 15) << 2;
    v4f f = *(const v4f*)&W[(k0 + kk) * E_ + tile_n + n4];
    Wt[n4 + 0][kk] = (bf16)f[0];
    Wt[n4 + 1][kk] = (bf16)f[1];
    Wt[n4 + 2][kk] = (bf16)f[2];
    Wt[n4 + 3][kk] = (bf16)f[3];
  }
}

__device__ __forceinline__ void stage_kv(const bf16* __restrict__ Kb,
                                         const bf16* __restrict__ Vb,
                                         bf16 (*Ks)[72], bf16 (*Vt)[72],
                                         int j0, int tid) {
#pragma unroll
  for (int it = 0; it < 4; ++it) {
    int v = tid + it * 128;
    int r = v >> 3, d8 = (v & 7) << 3;
    *(v8bf*)&Ks[r][d8] = *(const v8bf*)&Kb[(j0 + r) * DH_ + d8];
  }
#pragma unroll
  for (int it = 0; it < 4; ++it) {
    int v = tid + it * 128;
    int r = v >> 3, d8 = (v & 7) << 3;
    v8bf val = *(const v8bf*)&Vb[(j0 + r) * DH_ + d8];
#pragma unroll
    for (int i = 0; i < 8; ++i) Vt[d8 + i][r] = val[i];
  }
}

// ---- Kernel 1: fused Q/K~/V projections --------------------------------
// mode 0: Q = q@Wq ; mode 1: K~ = (kv@Wk) with d -> (64-d)%64 ; mode 2: V = kv@Wv
__global__ __launch_bounds__(128) void proj_kernel(
    const float* __restrict__ q, const float* __restrict__ kv,
    const float* __restrict__ Wq, const float* __restrict__ Wk,
    const float* __restrict__ Wv,
    bf16* __restrict__ Qws, bf16* __restrict__ Kws, bf16* __restrict__ Vws) {
  __shared__ bf16 Xs[2][64][40];
  __shared__ bf16 Wt[2][64][40];
  const int mode = blockIdx.y;
  const float* X = (mode == 0) ? q : kv;
  const float* W = (mode == 0) ? Wq : (mode == 1) ? Wk : Wv;
  bf16* dst = (mode == 0) ? Qws : (mode == 1) ? Kws : Vws;
  const int tile_m = (blockIdx.x >> 3) << 6;
  const int tile_n = (blockIdx.x & 7) << 6;
  const int tid = threadIdx.x, lane = tid & 31, wave = tid >> 5;
  const int wm = (wave >> 1) << 5, wn = (wave & 1) << 5;

  v8f c00 = {}, c01 = {}, c10 = {}, c11 = {};
  stage_xw(X, W, Xs[0], Wt[0], tile_m, tile_n, 0, tid);
  __syncthreads();
  for (int kb = 0; kb < 16; ++kb) {
    const int cur = kb & 1;
    if (kb < 15)
      stage_xw(X, W, Xs[cur ^ 1], Wt[cur ^ 1], tile_m, tile_n, (kb + 1) << 5, tid);
    v16bf a0 = frag_A(&Xs[cur][0][0], 40, wm,      0, lane);
    v16bf a1 = frag_A(&Xs[cur][0][0], 40, wm + 16, 0, lane);
    v16bf b0 = frag_B(&Wt[cur][0][0], 40, wn,      0, lane);
    v16bf b1 = frag_B(&Wt[cur][0][0], 40, wn + 16, 0, lane);
    c00 = WMMA_BF16(a0, b0, c00);
    c01 = WMMA_BF16(a0, b1, c01);
    c10 = WMMA_BF16(a1, b0, c10);
    c11 = WMMA_BF16(a1, b1, c11);
    __syncthreads();
  }
  const int permute_d = (mode == 1);
#pragma unroll
  for (int tm = 0; tm < 2; ++tm) {
#pragma unroll
    for (int tn = 0; tn < 2; ++tn) {
      v8f cv = (tm == 0) ? ((tn == 0) ? c00 : c01) : ((tn == 0) ? c10 : c11);
#pragma unroll
      for (int v = 0; v < 8; ++v) {
        int r = tile_m + wm + tm * 16 + v + ((lane >> 4) << 3);
        int c = tile_n + wn + tn * 16 + (lane & 15);
        int b = r >> 10, t = r & (T_ - 1);
        int h = c & (H_ - 1), d0 = c >> 3;
        int d = permute_d ? ((DH_ - d0) & (DH_ - 1)) : d0;
        dst[(((b << 3) + h) * T_ + t) * DH_ + d] = (bf16)cv[v];
      }
    }
  }
}

// ---- Kernel 2: flash-style attention, one (b,h) x 64 query rows per block
__global__ __launch_bounds__(128) void attn_kernel(
    const bf16* __restrict__ Qws, const bf16* __restrict__ Kws,
    const bf16* __restrict__ Vws, bf16* __restrict__ Mws) {
  __shared__ bf16 Qs[64][72];
  __shared__ bf16 Ks[2][64][72];    // [j][d] == B-transposed layout for QK^T
  __shared__ bf16 Vt[2][64][72];    // [d][j] == B-transposed layout for PV
  __shared__ bf16 Ps[4][16][72];    // per-wave P round-trip (wave-private)
  const int bh = blockIdx.x >> 4;
  const int i0 = (blockIdx.x & 15) << 6;
  const int tid = threadIdx.x, lane = tid & 31, wave = tid >> 5;
  const bf16* Qb = Qws + bh * (T_ * DH_);
  const bf16* Kb = Kws + bh * (T_ * DH_);
  const bf16* Vb = Vws + bh * (T_ * DH_);

#pragma unroll
  for (int it = 0; it < 4; ++it) {
    int v = tid + it * 128;
    int r = v >> 3, d8 = (v & 7) << 3;
    *(v8bf*)&Qs[r][d8] = *(const v8bf*)&Qb[(i0 + r) * DH_ + d8];
  }
  stage_kv(Kb, Vb, Ks[0], Vt[0], 0, tid);

  float m[8], l[8];
#pragma unroll
  for (int v = 0; v < 8; ++v) { m[v] = -1e30f; l[v] = 0.f; }
  v8f o0 = {}, o1 = {}, o2 = {}, o3 = {};
  __syncthreads();

  for (int jb = 0; jb < 16; ++jb) {
    const int cur = jb & 1;
    if (jb < 15)
      stage_kv(Kb, Vb, Ks[cur ^ 1], Vt[cur ^ 1], (jb + 1) << 6, tid);

    v8f s[4] = {v8f{}, v8f{}, v8f{}, v8f{}};
#pragma unroll
    for (int k0 = 0; k0 < DH_; k0 += 32) {
      v16bf aq = frag_A(&Qs[0][0], 72, wave << 4, k0, lane);
#pragma unroll
      for (int jt = 0; jt < 4; ++jt) {
        v16bf bk = frag_B(&Ks[cur][0][0], 72, jt << 4, k0, lane);
        s[jt] = WMMA_BF16(aq, bk, s[jt]);
      }
    }
    // online softmax: VGPR v holds row (v + 8*(lane>=16)), col = 16*jt + lane%16
#pragma unroll
    for (int v = 0; v < 8; ++v) {
      float tmax = -1e30f;
#pragma unroll
      for (int jt = 0; jt < 4; ++jt) { s[jt][v] *= SCALE_; tmax = fmaxf(tmax, s[jt][v]); }
#pragma unroll
      for (int msk = 1; msk < 16; msk <<= 1) tmax = fmaxf(tmax, __shfl_xor(tmax, msk, 32));
      float mnew = fmaxf(m[v], tmax);
      float corr = __expf(m[v] - mnew);
      float rsum = 0.f;
#pragma unroll
      for (int jt = 0; jt < 4; ++jt) {
        float p = __expf(s[jt][v] - mnew);
        s[jt][v] = p;
        rsum += p;
      }
#pragma unroll
      for (int msk = 1; msk < 16; msk <<= 1) rsum += __shfl_xor(rsum, msk, 32);
      l[v] = l[v] * corr + rsum;
      m[v] = mnew;
      o0[v] *= corr; o1[v] *= corr; o2[v] *= corr; o3[v] *= corr;
    }
    // spill P to LDS (C-layout -> A-layout reshape); Ps[wave] is wave-private,
    // and same-wave DS ops are in-order, so no barrier is needed before reload.
    {
      int pr = (lane >> 4) << 3, pc = lane & 15;
#pragma unroll
      for (int v = 0; v < 8; ++v)
#pragma unroll
        for (int jt = 0; jt < 4; ++jt)
          Ps[wave][v + pr][(jt << 4) + pc] = (bf16)s[jt][v];
    }
#pragma unroll
    for (int k0 = 0; k0 < 64; k0 += 32) {
      v16bf ap = frag_A(&Ps[wave][0][0], 72, 0, k0, lane);
      v16bf b0 = frag_B(&Vt[cur][0][0], 72,  0, k0, lane);
      v16bf b1 = frag_B(&Vt[cur][0][0], 72, 16, k0, lane);
      v16bf b2 = frag_B(&Vt[cur][0][0], 72, 32, k0, lane);
      v16bf b3 = frag_B(&Vt[cur][0][0], 72, 48, k0, lane);
      o0 = WMMA_BF16(ap, b0, o0);
      o1 = WMMA_BF16(ap, b1, o1);
      o2 = WMMA_BF16(ap, b2, o2);
      o3 = WMMA_BF16(ap, b3, o3);
    }
    __syncthreads();
  }
  // epilogue: /l, write merged 'b t (h d)' bf16
  const int b = bh >> 3, h = bh & 7;
#pragma unroll
  for (int v = 0; v < 8; ++v) {
    float inv = 1.0f / l[v];
    int r = i0 + (wave << 4) + v + ((lane >> 4) << 3);
    bf16* outp = Mws + (size_t)(b * T_ + r) * E_ + h * DH_ + (lane & 15);
    outp[ 0] = (bf16)(o0[v] * inv);
    outp[16] = (bf16)(o1[v] * inv);
    outp[32] = (bf16)(o2[v] * inv);
    outp[48] = (bf16)(o3[v] * inv);
  }
}

// ---- Kernel 3: out = merged @ Wo (fp32 out) -----------------------------
__global__ __launch_bounds__(128) void outproj_kernel(
    const bf16* __restrict__ Mws, const float* __restrict__ Wo,
    float* __restrict__ out) {
  __shared__ bf16 Xs[2][64][40];
  __shared__ bf16 Wt[2][64][40];
  const int tile_m = (blockIdx.x >> 3) << 6;
  const int tile_n = (blockIdx.x & 7) << 6;
  const int tid = threadIdx.x, lane = tid & 31, wave = tid >> 5;
  const int wm = (wave >> 1) << 5, wn = (wave & 1) << 5;

  auto stage = [&](bf16 (*Xd)[40], bf16 (*Wd)[40], int k0) {
#pragma unroll
    for (int it = 0; it < 2; ++it) {
      int v = tid + it * 128;
      int r = v >> 2, c8 = (v & 3) << 3;
      *(v8bf*)&Xd[r][c8] = *(const v8bf*)&Mws[(size_t)(tile_m + r) * E_ + k0 + c8];
    }
#pragma unroll
    for (int it = 0; it < 4; ++it) {
      int idx = tid + it * 128;
      int kk = idx >> 4, n4 = (idx & 15) << 2;
      v4f f = *(const v4f*)&Wo[(k0 + kk) * E_ + tile_n + n4];
      Wd[n4 + 0][kk] = (bf16)f[0];
      Wd[n4 + 1][kk] = (bf16)f[1];
      Wd[n4 + 2][kk] = (bf16)f[2];
      Wd[n4 + 3][kk] = (bf16)f[3];
    }
  };

  v8f c00 = {}, c01 = {}, c10 = {}, c11 = {};
  stage(Xs[0], Wt[0], 0);
  __syncthreads();
  for (int kb = 0; kb < 16; ++kb) {
    const int cur = kb & 1;
    if (kb < 15) stage(Xs[cur ^ 1], Wt[cur ^ 1], (kb + 1) << 5);
    v16bf a0 = frag_A(&Xs[cur][0][0], 40, wm,      0, lane);
    v16bf a1 = frag_A(&Xs[cur][0][0], 40, wm + 16, 0, lane);
    v16bf b0 = frag_B(&Wt[cur][0][0], 40, wn,      0, lane);
    v16bf b1 = frag_B(&Wt[cur][0][0], 40, wn + 16, 0, lane);
    c00 = WMMA_BF16(a0, b0, c00);
    c01 = WMMA_BF16(a0, b1, c01);
    c10 = WMMA_BF16(a1, b0, c10);
    c11 = WMMA_BF16(a1, b1, c11);
    __syncthreads();
  }
#pragma unroll
  for (int tm = 0; tm < 2; ++tm) {
#pragma unroll
    for (int tn = 0; tn < 2; ++tn) {
      v8f cv = (tm == 0) ? ((tn == 0) ? c00 : c01) : ((tn == 0) ? c10 : c11);
#pragma unroll
      for (int v = 0; v < 8; ++v) {
        int r = tile_m + wm + tm * 16 + v + ((lane >> 4) << 3);
        int c = tile_n + wn + tn * 16 + (lane & 15);
        out[(size_t)r * E_ + c] = cv[v];
      }
    }
  }
}

extern "C" void kernel_launch(void* const* d_in, const int* in_sizes, int n_in,
                              void* d_out, int out_size, void* d_ws, size_t ws_size,
                              hipStream_t stream) {
  const float* q  = (const float*)d_in[0];
  const float* kv = (const float*)d_in[1];
  const float* Wq = (const float*)d_in[2];
  const float* Wk = (const float*)d_in[3];
  const float* Wv = (const float*)d_in[4];
  const float* Wo = (const float*)d_in[5];
  // d_in[6] = head_num (constant 8, baked in)

  char* ws = (char*)d_ws;
  bf16* Qws = (bf16*)(ws);
  bf16* Kws = (bf16*)(ws + (4u << 20));
  bf16* Vws = (bf16*)(ws + (8u << 20));
  bf16* Mws = (bf16*)(ws + (12u << 20));

  dim3 blk(128);
  proj_kernel<<<dim3(512, 3), blk, 0, stream>>>(q, kv, Wq, Wk, Wv, Qws, Kws, Vws);
  attn_kernel<<<dim3(512), blk, 0, stream>>>(Qws, Kws, Vws, Mws);
  outproj_kernel<<<dim3(512), blk, 0, stream>>>(Mws, Wo, (float*)d_out);
}
